// HierarchicalJointModelBERT_61959198212178
// MI455X (gfx1250) — matrix operand order repaired and checked by reading, not compiled, not run on hardware
//
#include <hip/hip_runtime.h>
#include <math.h>

// ---------------------------------------------------------------------------
// CDNA5 (gfx1250) WMMA types & helpers
// ---------------------------------------------------------------------------
typedef __attribute__((ext_vector_type(16))) __bf16 v16bf;
typedef __attribute__((ext_vector_type(8)))  float  v8f;

union BF16x16 { v16bf v; unsigned short u[16]; };

__device__ __forceinline__ unsigned short f2bf(float f) {
    unsigned u = __float_as_uint(f);
    return (unsigned short)((u + 0x7FFFu + ((u >> 16) & 1u)) >> 16);   // RNE
}
__device__ __forceinline__ float bf2f(unsigned short u) {
    return __uint_as_float(((unsigned)u) << 16);
}
__device__ __forceinline__ v8f wmma_bf16(v16bf a, v16bf b, v8f c) {
    // (neg_a, A, neg_b, B, c_mod, C, reuse_a, reuse_b)
    return __builtin_amdgcn_wmma_f32_16x16x32_bf16(false, a, false, b, (short)0, c, false, false);
}
__device__ __forceinline__ v8f vzero8() { v8f a; for (int i = 0; i < 8; ++i) a[i] = 0.f; return a; }
__device__ __forceinline__ float sigf(float x) { return 1.f / (1.f + expf(-x)); }

// Model dims
#define DP 8
#define DR 10
#define DS 20
#define DT 30
#define DU 20
#define DLBL 9
#define NROWS (DP*DR*DS*DT)          // 48000 projected token rows
#define NSENT (DP*DR*DS)             // 1600 sentences

// ---------------------------------------------------------------------------
// generic f32 -> bf16 convert / zero
// ---------------------------------------------------------------------------
__global__ void k_cvt(const float* __restrict__ s, unsigned short* __restrict__ d, int n) {
    int i = blockIdx.x * blockDim.x + threadIdx.x;
    if (i < n) d[i] = f2bf(s[i]);
}
__global__ void k_zero(unsigned* __restrict__ p, int n) {
    int i = blockIdx.x * blockDim.x + threadIdx.x;
    if (i < n) p[i] = 0u;
}
// conv weights: fused cvt + WMMA B-fragment repack: pk[ks][lane][16]
__global__ void k_packB(const float* __restrict__ Wc, unsigned short* __restrict__ pk, int kst) {
    int e = blockIdx.x * blockDim.x + threadIdx.x;
    if (e >= kst * 512) return;
    int ks = e >> 9, l = (e >> 4) & 31, i = e & 15;
    int k = ks * 32 + ((l >> 4) << 4) + i;
    pk[e] = f2bf(Wc[k * 16 + (l & 15)]);
}

// ---------------------------------------------------------------------------
// Projection GEMM, M-tile = 32:  [48000,768] @ Wb.T + bias  (bf16 WMMA)
// Fused epilogue: per-row sums (tok_mask) reduced in LDS; output stored once
// as pre-masked bf16 (feeds the CNN); row sums stored for sentence mask.
// block = 256 thr (8 waves); wave = 2 M-subtiles x 2 N-subtiles.
// ---------------------------------------------------------------------------
__global__ __launch_bounds__(256) void k_proj(const float* __restrict__ X,
                                              const unsigned short* __restrict__ Wb,  // bf16 [256,768]
                                              const float* __restrict__ bias,
                                              unsigned short* __restrict__ Xb,        // bf16 [48000,256] masked
                                              float* __restrict__ rowsum_g) {
    __shared__ __align__(16) unsigned short As[32 * 768];
    __shared__ float rsum[32];
    const int m0 = blockIdx.x * 32;
    const int tid = threadIdx.x;
    if (tid < 32) rsum[tid] = 0.f;
#pragma unroll
    for (int i = 0; i < 24; ++i) {                         // 6144 float4 / 256 thr
        int idx = tid + i * 256;
        int row = idx / 192;
        int c4  = (idx % 192) * 4;
        const float4 v = *(const float4*)(X + (size_t)(m0 + row) * 768 + c4);
        unsigned short* d = &As[row * 768 + c4];
        d[0] = f2bf(v.x); d[1] = f2bf(v.y); d[2] = f2bf(v.z); d[3] = f2bf(v.w);
    }
    if (m0 + 32 < NROWS && tid < 192)                      // warm next A tile (global_prefetch_b8)
        __builtin_prefetch(X + (size_t)(m0 + 32) * 768 + tid * 128, 0, 1);
    __syncthreads();

    const int wave = tid >> 5, lane = tid & 31;
    const int lr = lane & 15, hi = lane >> 4;
    const int n0a = wave * 32, n0b = wave * 32 + 16;
    v8f acc[2][2];
#pragma unroll
    for (int mi = 0; mi < 2; ++mi) { acc[mi][0] = vzero8(); acc[mi][1] = vzero8(); }

    for (int ks = 0; ks < 24; ++ks) {
        const int k0 = ks * 32;
        BF16x16 b0, b1;
        const unsigned short* bp0 = Wb + (size_t)(n0a + lr) * 768 + k0 + hi * 16;
        const unsigned short* bp1 = Wb + (size_t)(n0b + lr) * 768 + k0 + hi * 16;
        *(uint4*)&b0.u[0] = *(const uint4*)bp0; *(uint4*)&b0.u[8] = *(const uint4*)(bp0 + 8);
        *(uint4*)&b1.u[0] = *(const uint4*)bp1; *(uint4*)&b1.u[8] = *(const uint4*)(bp1 + 8);
#pragma unroll
        for (int mi = 0; mi < 2; ++mi) {
            BF16x16 a;
            const unsigned short* ap = &As[(mi * 16 + lr) * 768 + k0 + hi * 16];
            *(uint4*)&a.u[0] = *(const uint4*)ap;          // K base..base+7
            *(uint4*)&a.u[8] = *(const uint4*)(ap + 8);    // K base+8..base+15
            acc[mi][0] = wmma_bf16(a.v, b0.v, acc[mi][0]);
            acc[mi][1] = wmma_bf16(a.v, b1.v, acc[mi][1]);
        }
    }
    const float ba = bias[n0a + lr], bb = bias[n0b + lr];
    // row-sum reduction: shfl within each 16-lane half, then LDS atomic per row
#pragma unroll
    for (int mi = 0; mi < 2; ++mi) {
#pragma unroll
        for (int v = 0; v < 8; ++v) {
            float s = (acc[mi][0][v] + ba) + (acc[mi][1][v] + bb);
            for (int off = 8; off; off >>= 1) s += __shfl_xor(s, off);
            if (lr == 0) atomicAdd(&rsum[mi * 16 + v + hi * 8], s);
        }
    }
    __syncthreads();
    if (tid < 32) rowsum_g[m0 + tid] = rsum[tid];
#pragma unroll
    for (int mi = 0; mi < 2; ++mi) {
#pragma unroll
        for (int v = 0; v < 8; ++v) {
            const int rl = mi * 16 + v + hi * 8;
            const float m = (rsum[rl] != 0.f) ? 1.f : 0.f;   // tok_mask
            const size_t base = (size_t)(m0 + rl) * 256;
            Xb[base + n0a + lr] = f2bf((acc[mi][0][v] + ba) * m);
            Xb[base + n0b + lr] = f2bf((acc[mi][1][v] + bb) * m);
        }
    }
}

__global__ void k_sentmask(const float* __restrict__ rowsum, float* __restrict__ sm) {
    int s = blockIdx.x * blockDim.x + threadIdx.x;
    if (s < NSENT) {
        float t = 0.f;
        for (int i = 0; i < DT; ++i) t += rowsum[s * DT + i];
        sm[s] = (t != 0.f) ? 1.f : 0.f;
    }
}

// ---------------------------------------------------------------------------
// CNN conv as implicit-im2col WMMA GEMM over pre-masked bf16 activations.
// Conv window is contiguous in memory -> A frag = two 16B loads. B frags come
// pre-packed from global (L2 resident). relu+max via uint atomicMax (vals>=0).
// block = 128 thr (4 waves), wave = one 16-row M-tile; N = 16 (one subtile).
// ---------------------------------------------------------------------------
__global__ __launch_bounds__(128) void k_cnn(const unsigned short* __restrict__ Xb, // bf16 [48000,256]
                                             const unsigned short* __restrict__ pk, // packed B frags
                                             const float* __restrict__ bias,        // [16]
                                             unsigned* __restrict__ feats,          // [1600*64] uint
                                             int ksz, int foff) {
    const int kst = ksz * 8;                               // K-steps of 32
    const int tid = threadIdx.x;
    const int wave = tid >> 5, lane = tid & 31, lr = lane & 15, hi = lane >> 4;
    const int Wp = 31 - ksz;
    const int m0 = (blockIdx.x * 4 + wave) * 16;
    const int arow = m0 + lr;
    const int sent = arow / Wp, pos = arow % Wp;
    const unsigned short* xrow = Xb + (size_t)(sent * DT + pos) * 256;

    v8f acc = vzero8();
    for (int ks = 0; ks < kst; ++ks) {
        const int kbase = ks * 32 + hi * 16;
        BF16x16 a;
        *(uint4*)&a.u[0] = *(const uint4*)(xrow + kbase);
        *(uint4*)&a.u[8] = *(const uint4*)(xrow + kbase + 8);
        BF16x16 b;
        const unsigned short* bp = pk + (ks * 32 + lane) * 16;
        *(uint4*)&b.u[0] = *(const uint4*)bp; *(uint4*)&b.u[8] = *(const uint4*)(bp + 8);
        acc = wmma_bf16(a.v, b.v, acc);
    }
    const float bz = bias[lr];
#pragma unroll
    for (int v = 0; v < 8; ++v) {
        int row = m0 + v + hi * 8;
        int s2 = row / Wp;
        float val = acc[v] + bz; val = val > 0.f ? val : 0.f;
        atomicMax(&feats[s2 * 64 + foff + lr], __float_as_uint(val));
    }
}

// ---------------------------------------------------------------------------
// gx GEMM for review GRU: G[1600,768] = F[1600,64] @ Wih.T + bih   (WMMA)
// ---------------------------------------------------------------------------
__global__ __launch_bounds__(256) void k_gx(const float* __restrict__ F,
                                            const unsigned short* __restrict__ Wb, // bf16 [768,64]
                                            const float* __restrict__ bias,
                                            float* __restrict__ G) {
    __shared__ __align__(16) unsigned short As[16 * 64];
    const int m0 = blockIdx.x * 16;
    const int tid = threadIdx.x;
#pragma unroll
    for (int i = 0; i < 4; ++i) {
        int idx = tid + i * 256;
        As[idx] = f2bf(F[(size_t)m0 * 64 + idx]);
    }
    __syncthreads();
    const int wave = tid >> 5, lane = tid & 31, lr = lane & 15, hi = lane >> 4;
#pragma unroll
    for (int i = 0; i < 6; ++i) {
        const int n0 = (wave * 6 + i) * 16;
        v8f acc = vzero8();
#pragma unroll
        for (int ks = 0; ks < 2; ++ks) {
            const int k0 = ks * 32;
            BF16x16 a;
            const unsigned short* ap = &As[lr * 64 + k0 + hi * 16];
            *(uint4*)&a.u[0] = *(const uint4*)ap; *(uint4*)&a.u[8] = *(const uint4*)(ap + 8);
            BF16x16 b;
            const unsigned short* bp = Wb + (size_t)(n0 + lr) * 64 + k0 + hi * 16;
            *(uint4*)&b.u[0] = *(const uint4*)bp; *(uint4*)&b.u[8] = *(const uint4*)(bp + 8);
            acc = wmma_bf16(a.v, b.v, acc);
        }
        const float bb = bias[n0 + lr];
#pragma unroll
        for (int v = 0; v < 8; ++v)
            G[(size_t)(m0 + v + hi * 8) * 768 + n0 + lr] = acc[v] + bb;
    }
}

// ---------------------------------------------------------------------------
// GRU recurrence. grid=2: forward / backward directions on separate WGPs.
// One workgroup (16 waves) per direction; h kept bf16 in LDS; per step the
// r/z/n gate pre-activations are computed by WMMA, then f32 gate math.
// BT = batch/16 (5 for review GRU, 1 for product GRU).
// ---------------------------------------------------------------------------
template <int BT>
__global__ __launch_bounds__(512) void k_gru(const float* __restrict__ G0, const float* __restrict__ G1,
                                             const unsigned short* __restrict__ W0,
                                             const unsigned short* __restrict__ W1,
                                             const float* __restrict__ bh0, const float* __restrict__ bh1,
                                             const float* __restrict__ maskf,   // [B*L] or null
                                             float* __restrict__ out,           // [validB,512]
                                             int L, int validB) {
    const int dir = blockIdx.x;                  // 0 = forward, 1 = backward
    const float* G = dir ? G1 : G0;
    const unsigned short* Whh = dir ? W1 : W0;
    const float* bhh = dir ? bh1 : bh0;
    const int coloff = dir * 256;

    __shared__ __align__(16) unsigned short h[BT * 16 * 256];
    const int tid = threadIdx.x;
    for (int e = tid; e < BT * 16 * 256; e += 512) h[e] = 0;
    __syncthreads();

    const int wave = tid >> 5, lane = tid & 31, lr = lane & 15, hi = lane >> 4;
    const int j0 = wave * 16;                    // this wave owns columns j0..j0+15
    const int j  = j0 + lr;
    const float br = bhh[j], bz = bhh[j + 256], bn = bhh[j + 512];

    v8f acc[BT][3];
    for (int step = 0; step < L; ++step) {
        const int t = dir ? (L - 1 - step) : step;
#pragma unroll
        for (int mt = 0; mt < BT; ++mt)
#pragma unroll
            for (int g = 0; g < 3; ++g) acc[mt][g] = vzero8();

        for (int ks = 0; ks < 8; ++ks) {
            const int k0 = ks * 32;
            BF16x16 b[3];
#pragma unroll
            for (int g = 0; g < 3; ++g) {
                const unsigned short* bp = Whh + (size_t)(j0 + g * 256 + lr) * 256 + k0 + hi * 16;
                *(uint4*)&b[g].u[0] = *(const uint4*)bp; *(uint4*)&b[g].u[8] = *(const uint4*)(bp + 8);
            }
#pragma unroll
            for (int mt = 0; mt < BT; ++mt) {
                BF16x16 a;
                const unsigned short* ap = &h[(mt * 16 + lr) * 256 + k0 + hi * 16];
                *(uint4*)&a.u[0] = *(const uint4*)ap; *(uint4*)&a.u[8] = *(const uint4*)(ap + 8);
#pragma unroll
                for (int g = 0; g < 3; ++g)
                    acc[mt][g] = wmma_bf16(a.v, b[g].v, acc[mt][g]);
            }
        }
        __syncthreads();                 // all waves done reading h

#pragma unroll
        for (int mt = 0; mt < BT; ++mt) {
#pragma unroll
            for (int v = 0; v < 8; ++v) {
                const int b_ = mt * 16 + v + hi * 8;
                const float m = maskf ? maskf[b_ * L + t] : ((b_ < validB) ? 1.f : 0.f);
                const float* gp = G + (size_t)(b_ * L + t) * 768;
                const float r = sigf(gp[j]       + acc[mt][0][v] + br);
                const float z = sigf(gp[j + 256] + acc[mt][1][v] + bz);
                const float n = tanhf(gp[j + 512] + r * (acc[mt][2][v] + bn));
                const float hp = bf2f(h[b_ * 256 + j]);
                const float hn = (m != 0.f) ? ((1.f - z) * n + z * hp) : hp;
                h[b_ * 256 + j] = f2bf(hn);
            }
        }
        __syncthreads();
    }
    for (int e = tid; e < validB * 256; e += 512) {
        int b_ = e / 256, jj = e % 256;
        out[(size_t)b_ * 512 + coloff + jj] = bf2f(h[b_ * 256 + jj]);
    }
}

// ---------------------------------------------------------------------------
// user-feat L2 normalization (over R axis) + pb = concat(r_vec, ufn*ufw)
// ---------------------------------------------------------------------------
__global__ __launch_bounds__(256) void k_pb(const float* __restrict__ rvec,   // [80,512]
                                            const float* __restrict__ uf,     // [8,10,20]
                                            const float* __restrict__ ufw,    // [20]
                                            float* __restrict__ pb) {         // [80,532]
    __shared__ float inv[DP * DU];
    const int tid = threadIdx.x;
    if (tid < DP * DU) {
        int p = tid / DU, u = tid % DU;
        float s = 0.f;
        for (int r = 0; r < DR; ++r) { float v = uf[(p * DR + r) * DU + u]; s += v * v; }
        float n = sqrtf(s); n = n > 1e-12f ? n : 1e-12f;
        inv[tid] = 1.f / n;
    }
    __syncthreads();
    for (int e = tid; e < 80 * 532; e += 256) {
        int row = e / 532, c = e % 532;
        float v;
        if (c < 512) v = rvec[row * 512 + c];
        else {
            int u = c - 512, p = row / DR;
            v = uf[row * DU + u] * inv[p * DU + u] * ufw[u];
        }
        pb[e] = v;
    }
}

// ---------------------------------------------------------------------------
// product-GRU gx (VALU, tiny): G[(16*10),768], rows >= 80 zero-padded.
// ---------------------------------------------------------------------------
__global__ void k_gxp(const float* __restrict__ pb, const float* __restrict__ Wih, // [768,532]
                      const float* __restrict__ bih, float* __restrict__ G) {
    int o = blockIdx.x * blockDim.x + threadIdx.x;
    if (o >= 160 * 768) return;
    int row = o / 768, jc = o % 768;
    float s = 0.f;
    if (row / DR < DP) {
        const float* a = pb + (size_t)row * 532;
        const float* w = Wih + (size_t)jc * 532;
        s = bih[jc];
        for (int k = 0; k < 532; ++k) s += a[k] * w[k];
    }
    G[o] = s;
}

// ---------------------------------------------------------------------------
// review FF head -> r_stars (d_out + 72) ;  product FF head -> p_stars (d_out)
// ---------------------------------------------------------------------------
__global__ __launch_bounds__(128) void k_rff(const float* __restrict__ pb,
                                             const float* __restrict__ W1, const float* __restrict__ b1,
                                             const float* __restrict__ W2, const float* __restrict__ b2,
                                             float* __restrict__ out) {
    __shared__ float hb[128];
    const int row = blockIdx.x, jt = threadIdx.x;
    float s = b1[jt];
    const float* a = pb + (size_t)row * 532;
    const float* w = W1 + (size_t)jt * 532;
    for (int k = 0; k < 532; ++k) s += a[k] * w[k];
    s = s > 0.f ? s : expm1f(s);                           // elu
    hb[jt] = s * W2[jt];
    __syncthreads();
    for (int off = 64; off; off >>= 1) { if (jt < off) hb[jt] += hb[jt + off]; __syncthreads(); }
    if (jt == 0) out[row] = hb[0] + b2[0];
}
__global__ __launch_bounds__(128) void k_pff(const float* __restrict__ hcat,
                                             const float* __restrict__ W1, const float* __restrict__ b1,
                                             const float* __restrict__ W2, const float* __restrict__ b2,
                                             float* __restrict__ out) {
    __shared__ float g[128];
    const int row = blockIdx.x, jt = threadIdx.x;
    float s = b1[jt];
    const float* a = hcat + (size_t)row * 512;
    const float* w = W1 + (size_t)jt * 512;
    for (int k = 0; k < 512; ++k) s += a[k] * w[k];
    g[jt] = s > 0.f ? s : expm1f(s);
    __syncthreads();
    if (jt < DLBL) {
        float o = b2[jt];
        const float* w2 = W2 + jt * 128;
        for (int k = 0; k < 128; ++k) o += g[k] * w2[k];
        out[row * DLBL + jt] = o;
    }
}

// ---------------------------------------------------------------------------
// host side
// ---------------------------------------------------------------------------
struct ParamPtrs {
    const float *inputs, *user_feats;
    const float *convW[4], *convB[4];
    const float *pff1W, *pff1b, *pff2W, *pff2b;
    const float *pgbWhh, *pgbWih, *pgbbhh, *pgbbih;
    const float *pgfWhh, *pgfWih, *pgfbhh, *pgfbih;
    const float *projW, *projb;
    const float *rff1W, *rff1b, *rff2W, *rff2b;
    const float *rgbWhh, *rgbWih, *rgbbhh, *rgbbih;
    const float *rgfWhh, *rgfWih, *rgfbhh, *rgfbih;
    const float *ufw;
};

extern "C" void kernel_launch(void* const* d_in, const int* in_sizes, int n_in,
                              void* d_out, int out_size, void* d_ws, size_t ws_size,
                              hipStream_t stream) {
    auto f = [&](int i) { return (const float*)d_in[i]; };
    ParamPtrs p;
    if (n_in >= 2 && in_sizes[1] == 1600) {
        // dict insertion order: inputs, user_feats, proj, convs, rgru_f, rgru_b,
        // pgru_f, pgru_b, rff1, rff2, pff1, pff2, ufw  (W,b / Wih,Whh,bih,bhh)
        p.inputs = f(0); p.user_feats = f(1);
        p.projW = f(2); p.projb = f(3);
        for (int i = 0; i < 4; ++i) { p.convW[i] = f(4 + 2 * i); p.convB[i] = f(5 + 2 * i); }
        p.rgfWih = f(12); p.rgfWhh = f(13); p.rgfbih = f(14); p.rgfbhh = f(15);
        p.rgbWih = f(16); p.rgbWhh = f(17); p.rgbbih = f(18); p.rgbbhh = f(19);
        p.pgfWih = f(20); p.pgfWhh = f(21); p.pgfbih = f(22); p.pgfbhh = f(23);
        p.pgbWih = f(24); p.pgbWhh = f(25); p.pgbbih = f(26); p.pgbbhh = f(27);
        p.rff1W = f(28); p.rff1b = f(29); p.rff2W = f(30); p.rff2b = f(31);
        p.pff1W = f(32); p.pff1b = f(33); p.pff2W = f(34); p.pff2b = f(35);
        p.ufw = f(36);
    } else {
        // jax tree-flatten (sorted keys): inputs, params{convs, pff1, pff2, pgru_b,
        // pgru_f, proj, rff1, rff2, rgru_b, rgru_f, ufw}, user_feats; gru keys sorted
        // Whh, Wih, bhh, bih; lin keys W, b.
        p.inputs = f(0);
        for (int i = 0; i < 4; ++i) { p.convW[i] = f(1 + 2 * i); p.convB[i] = f(2 + 2 * i); }
        p.pff1W = f(9); p.pff1b = f(10); p.pff2W = f(11); p.pff2b = f(12);
        p.pgbWhh = f(13); p.pgbWih = f(14); p.pgbbhh = f(15); p.pgbbih = f(16);
        p.pgfWhh = f(17); p.pgfWih = f(18); p.pgfbhh = f(19); p.pgfbih = f(20);
        p.projW = f(21); p.projb = f(22);
        p.rff1W = f(23); p.rff1b = f(24); p.rff2W = f(25); p.rff2b = f(26);
        p.rgbWhh = f(27); p.rgbWih = f(28); p.rgbbhh = f(29); p.rgbbih = f(30);
        p.rgfWhh = f(31); p.rgfWih = f(32); p.rgfbhh = f(33); p.rgfbih = f(34);
        p.ufw = f(35); p.user_feats = f(36);
    }

    // workspace layout
    char* w = (char*)d_ws;
    size_t off = 0;
    auto alloc = [&](size_t bytes) -> char* {
        char* ptr = w + off; off += (bytes + 255) & ~(size_t)255; return ptr;
    };
    unsigned short* wp_bf  = (unsigned short*)alloc(196608 * 2);
    unsigned short* whh_rf = (unsigned short*)alloc(196608 * 2);
    unsigned short* whh_rb = (unsigned short*)alloc(196608 * 2);
    unsigned short* whh_pf = (unsigned short*)alloc(196608 * 2);
    unsigned short* whh_pb = (unsigned short*)alloc(196608 * 2);
    unsigned short* wih_rf = (unsigned short*)alloc(49152 * 2);
    unsigned short* wih_rb = (unsigned short*)alloc(49152 * 2);
    const int kszs[4] = {2, 3, 4, 5};
    unsigned short* pkB[4];
    for (int i = 0; i < 4; ++i) pkB[i] = (unsigned short*)alloc((size_t)(kszs[i] * 8) * 512 * 2);
    unsigned short* xbf = (unsigned short*)alloc((size_t)NROWS * 256 * 2);  // masked bf16 act.
    float* rowsum = (float*)alloc((size_t)NROWS * 4);
    float* sentm  = (float*)alloc((size_t)NSENT * 4);
    float* feats  = (float*)alloc((size_t)NSENT * 64 * 4);
    float* gx_f   = (float*)alloc((size_t)NSENT * 768 * 4);
    float* gx_b   = (float*)alloc((size_t)NSENT * 768 * 4);
    float* rvec   = (float*)alloc(80 * 512 * 4);
    float* pbuf   = (float*)alloc(80 * 532 * 4);
    float* gxp_f  = (float*)alloc(160 * 768 * 4);
    float* gxp_b  = (float*)alloc(160 * 768 * 4);
    float* hcat   = (float*)alloc(8 * 512 * 4);
    (void)ws_size; (void)out_size; (void)n_in;

    auto cvt = [&](const float* s, unsigned short* d, int n) {
        k_cvt<<<(n + 255) / 256, 256, 0, stream>>>(s, d, n);
    };
    cvt(p.projW,  wp_bf,  196608);
    cvt(p.rgfWhh, whh_rf, 196608); cvt(p.rgbWhh, whh_rb, 196608);
    cvt(p.pgfWhh, whh_pf, 196608); cvt(p.pgbWhh, whh_pb, 196608);
    cvt(p.rgfWih, wih_rf, 49152);  cvt(p.rgbWih, wih_rb, 49152);
    for (int i = 0; i < 4; ++i) {
        int kst = kszs[i] * 8;
        k_packB<<<(kst * 512 + 255) / 256, 256, 0, stream>>>(p.convW[i], pkB[i], kst);
    }
    k_zero<<<(NSENT * 64 + 255) / 256, 256, 0, stream>>>((unsigned*)feats, NSENT * 64);

    // 1) projection GEMM (WMMA) with fused masking -> bf16 activations + row sums
    k_proj<<<NROWS / 32, 256, 0, stream>>>(p.inputs, wp_bf, p.projb, xbf, rowsum);
    k_sentmask<<<(NSENT + 255) / 256, 256, 0, stream>>>(rowsum, sentm);

    // 2) CNN encoder (WMMA + atomic max); feats layout [sent][64], ksz blocks of 16
    for (int i = 0; i < 4; ++i) {
        int ksz = kszs[i];
        int grid = (NSENT * (31 - ksz)) / 64;
        k_cnn<<<grid, 128, 0, stream>>>(xbf, pkB[i], p.convB[i], (unsigned*)feats, ksz, i * 16);
    }

    // 3) review GRU: batched gx (WMMA), then sequential recurrence (WMMA),
    //    forward+backward concurrently on two WGPs (grid=2)
    k_gx<<<NSENT / 16, 256, 0, stream>>>(feats, wih_rf, p.rgfbih, gx_f);
    k_gx<<<NSENT / 16, 256, 0, stream>>>(feats, wih_rb, p.rgbbih, gx_b);
    k_gru<5><<<2, 512, 0, stream>>>(gx_f, gx_b, whh_rf, whh_rb, p.rgfbhh, p.rgbbhh,
                                    sentm, rvec, DS, 80);

    // 4) pb = [r_vec, normalized user feats * ufw]; review head -> r_stars
    k_pb<<<1, 256, 0, stream>>>(rvec, p.user_feats, p.ufw, pbuf);
    k_rff<<<80, 128, 0, stream>>>(pbuf, p.rff1W, p.rff1b, p.rff2W, p.rff2b,
                                  (float*)d_out + DP * DLBL);

    // 5) product GRU (batch padded 8->16, fwd+bwd concurrently) and product head
    k_gxp<<<(160 * 768 + 255) / 256, 256, 0, stream>>>(pbuf, p.pgfWih, p.pgfbih, gxp_f);
    k_gxp<<<(160 * 768 + 255) / 256, 256, 0, stream>>>(pbuf, p.pgbWih, p.pgbbih, gxp_b);
    k_gru<1><<<2, 512, 0, stream>>>(gxp_f, gxp_b, whh_pf, whh_pb, p.pgfbhh, p.pgbbhh,
                                    nullptr, hcat, DR, 8);
    k_pff<<<DP, 128, 0, stream>>>(hcat, p.pff1W, p.pff1b, p.pff2W, p.pff2b, (float*)d_out);
}
